// SpecializedMultiHeadAttention_81604378624402
// MI455X (gfx1250) — matrix operand order repaired and checked by reading, not compiled, not run on hardware
//
#include <hip/hip_runtime.h>
#include <hip/hip_bf16.h>

typedef unsigned short u16;
typedef unsigned int   u32;

typedef __attribute__((ext_vector_type(16))) __bf16 v16bf;
typedef __attribute__((ext_vector_type(8)))  float  v8f;

// Problem sizes (fixed by the reference)
constexpr int Bn = 8, Sn = 2048, Dn = 256, Hn = 4, HDn = 64;
constexpr int Mn = Bn * Sn;  // 16384 rows in the projection GEMMs

// ---------- helpers ----------

__device__ __forceinline__ u16 f2bf(float f) {
  // round-to-nearest-even fp32 -> bf16
  u32 u = __float_as_uint(f);
  u32 r = (u + 0x7FFFu + ((u >> 16) & 1u)) >> 16;
  return (u16)r;
}

struct Frag32B { uint4 a, b; };
struct U16x16  { u16 v[16]; };

// A-operand (16-bit, 16x32): lane 0-15 row M=lane, K chunks {0-7,16-23};
// lane 16-31 same rows, K chunks {8-15,24-31}.
__device__ __forceinline__ v16bf load_fragA(const u16* rowbase, int k0, int half) {
  const u16* p0 = rowbase + k0 + half * 8;
  Frag32B s;
  s.a = *(const uint4*)(p0);        // K = k0 + half*8 .. +7
  s.b = *(const uint4*)(p0 + 16);   // K = k0 + 16 + half*8 .. +7
  return __builtin_bit_cast(v16bf, s);
}

// B-operand (16-bit, 32x16): lane = column N, lanes 0-15 hold K chunks {0-7,16-23},
// lanes 16-31 hold K chunks {8-15,24-31} (same order as A).
__device__ __forceinline__ v16bf load_fragB(const u16* colbase, int k0, int half) {
  const u16* p = colbase + k0 + half * 8;
  Frag32B s;
  s.a = *(const uint4*)(p);
  s.b = *(const uint4*)(p + 16);
  return __builtin_bit_cast(v16bf, s);
}

// A-operand built from fp32 data (attention matrix), converted to bf16 in regs.
__device__ __forceinline__ v16bf fragA_from_f32(const float* rowbase, int k0, int half) {
  const float* p0 = rowbase + k0 + half * 8;
  const float* p1 = p0 + 16;
  const float4 f0 = *(const float4*)(p0);
  const float4 f1 = *(const float4*)(p0 + 4);
  const float4 f2 = *(const float4*)(p1);
  const float4 f3 = *(const float4*)(p1 + 4);
  U16x16 t;
  t.v[0] = f2bf(f0.x); t.v[1] = f2bf(f0.y); t.v[2]  = f2bf(f0.z); t.v[3]  = f2bf(f0.w);
  t.v[4] = f2bf(f1.x); t.v[5] = f2bf(f1.y); t.v[6]  = f2bf(f1.z); t.v[7]  = f2bf(f1.w);
  t.v[8] = f2bf(f2.x); t.v[9] = f2bf(f2.y); t.v[10] = f2bf(f2.z); t.v[11] = f2bf(f2.w);
  t.v[12]= f2bf(f3.x); t.v[13]= f2bf(f3.y); t.v[14] = f2bf(f3.z); t.v[15] = f2bf(f3.w);
  return __builtin_bit_cast(v16bf, t);
}

__device__ __forceinline__ v8f wmma_bf16(v16bf a, v16bf b, v8f c) {
  return __builtin_amdgcn_wmma_f32_16x16x32_bf16(false, a, false, b, (short)0, c,
                                                 false, false);
}

// ---------- kernels ----------

__global__ void f32_to_bf16_k(const float* __restrict__ in, u16* __restrict__ out, int n) {
  int i = blockIdx.x * blockDim.x + threadIdx.x;
  int stride = gridDim.x * blockDim.x;
  for (; i < n; i += stride) out[i] = f2bf(in[i]);
}

// y[m,n] = sum_k A[m,k] * W[n,k] + bias[n]; one wave per 16x16 output tile.
// mode 0: bf16 row-major [M,D]   (Q, K)
// mode 1: fp32 row-major [M,D]   (final output projection)
// mode 2: bf16 transposed-per-head [B,H,HD,S]  (V)
__global__ void gemm_proj(const u16* __restrict__ A, const u16* __restrict__ W,
                          const float* __restrict__ bias,
                          u16* __restrict__ outb, float* __restrict__ outf, int mode) {
  const int lane = threadIdx.x;
  const int half = lane >> 4;
  const int lr   = lane & 15;
  const int tm = blockIdx.x, tn = blockIdx.y;
  const u16* Arow = A + (size_t)(tm * 16 + lr) * Dn;
  const u16* Wrow = W + (size_t)(tn * 16 + lr) * Dn;
  v8f c = {};
#pragma unroll
  for (int k0 = 0; k0 < Dn; k0 += 32) {
    c = wmma_bf16(load_fragA(Arow, k0, half), load_fragB(Wrow, k0, half), c);
  }
  const int col = tn * 16 + lr;
  const float bv = bias[col];
#pragma unroll
  for (int r = 0; r < 8; ++r) {
    const int row = tm * 16 + r + half * 8;
    const float v = c[r] + bv;
    if (mode == 0) {
      outb[(size_t)row * Dn + col] = f2bf(v);
    } else if (mode == 1) {
      outf[(size_t)row * Dn + col] = v;
    } else {
      const int b_ = row / Sn, s_ = row % Sn;
      const int h_ = col / HDn, hd = col % HDn;
      outb[(((size_t)b_ * Hn + h_) * HDn + hd) * Sn + s_] = f2bf(v);
    }
  }
}

// scores tile = Q K^T / sqrt(HD) + per-head bias; lower-triangular tiles only.
__global__ void attn_scores(const u16* __restrict__ Qb, const u16* __restrict__ Kb,
                            const float* __restrict__ bowler,
                            const float* __restrict__ batsman,
                            const float* __restrict__ rec_s,
                            const float* __restrict__ bowl_s,
                            const float* __restrict__ bat_s,
                            float* __restrict__ attn) {
  const int tk = blockIdx.x, tq = blockIdx.y, bh = blockIdx.z;
  if (tk > tq) return;  // causal: softmax pass writes zeros above the diagonal
  const int b_ = bh / Hn, h_ = bh % Hn;
  const int lane = threadIdx.x, half = lane >> 4, lr = lane & 15;
  const u16* Qrow = Qb + ((size_t)b_ * Sn + tq * 16 + lr) * Dn + h_ * HDn;
  const u16* Krow = Kb + ((size_t)b_ * Sn + tk * 16 + lr) * Dn + h_ * HDn;
  v8f c = {};
#pragma unroll
  for (int k0 = 0; k0 < HDn; k0 += 32) {
    c = wmma_bf16(load_fragA(Qrow, k0, half), load_fragB(Krow, k0, half), c);
  }
  const int kcol = tk * 16 + lr;
  float hbias = 0.f;
  if (h_ == 0) hbias = rec_s[0] * ((float)kcol / (float)Sn);
  const float bw = (h_ == 1) ? bowl_s[0] : 0.f;
  const float bt = (h_ == 2) ? bat_s[0] : 0.f;
  float* abase = attn + (size_t)bh * Sn * Sn;
#pragma unroll
  for (int r = 0; r < 8; ++r) {
    const int q = tq * 16 + r + half * 8;
    float v = c[r] * 0.125f + hbias;  // 1/sqrt(64)
    if (h_ == 1) v += bw * bowler[((size_t)b_ * Sn + q) * Sn + kcol];
    if (h_ == 2) v += bt * batsman[((size_t)b_ * Sn + q) * Sn + kcol];
    abase[(size_t)q * Sn + kcol] = v;
  }
}

// In-place causal softmax over each [b,h,q] row; writes 0 for k > q.
__global__ void softmax_causal(float* __restrict__ attn) {
  const int q  = blockIdx.x;
  const int bh = blockIdx.y;
  float* row = attn + ((size_t)bh * Sn + q) * Sn;
  const int t = threadIdx.x;
  const int nvalid = q + 1;
  __shared__ float red[256];
  float lmax = -3.4e38f;
  for (int k = t; k < nvalid; k += 256) lmax = fmaxf(lmax, row[k]);
  red[t] = lmax; __syncthreads();
  for (int s = 128; s > 0; s >>= 1) {
    if (t < s) red[t] = fmaxf(red[t], red[t + s]);
    __syncthreads();
  }
  const float m = red[0]; __syncthreads();
  float lsum = 0.f;
  for (int k = t; k < nvalid; k += 256) lsum += __expf(row[k] - m);
  red[t] = lsum; __syncthreads();
  for (int s = 128; s > 0; s >>= 1) {
    if (t < s) red[t] += red[t + s];
    __syncthreads();
  }
  const float inv = 1.0f / red[0];
  for (int k = t; k < Sn; k += 256)
    row[k] = (k < nvalid) ? __expf(row[k] - m) * inv : 0.f;
}

// out[q,hd] = sum_k attn[q,k] * V[k,hd]; K-loop stops at the causal diagonal.
__global__ void attn_v(const float* __restrict__ attn, const u16* __restrict__ Vt,
                       u16* __restrict__ attnOut) {
  const int tq = blockIdx.x, thd = blockIdx.y, bh = blockIdx.z;
  const int b_ = bh / Hn, h_ = bh % Hn;
  const int lane = threadIdx.x, half = lane >> 4, lr = lane & 15;
  const float* Arow = attn + ((size_t)bh * Sn + tq * 16 + lr) * Sn;
  const u16* Vrow = Vt + ((size_t)bh * HDn + thd * 16 + lr) * Sn;
  v8f c = {};
  const int kend = (tq + 1) * 16;  // keys beyond diagonal are exactly zero
  for (int k0 = 0; k0 < kend; k0 += 32) {
    c = wmma_bf16(fragA_from_f32(Arow, k0, half), load_fragB(Vrow, k0, half), c);
  }
  const int dcol = h_ * HDn + thd * 16 + lr;
#pragma unroll
  for (int r = 0; r < 8; ++r) {
    const int q = tq * 16 + r + half * 8;
    attnOut[((size_t)b_ * Sn + q) * Dn + dcol] = f2bf(c[r]);
  }
}

// ---------- launch ----------

extern "C" void kernel_launch(void* const* d_in, const int* in_sizes, int n_in,
                              void* d_out, int out_size, void* d_ws, size_t ws_size,
                              hipStream_t stream) {
  const float* x       = (const float*)d_in[0];
  const float* bowler  = (const float*)d_in[1];
  const float* batsman = (const float*)d_in[2];
  const float* Wq = (const float*)d_in[3];  const float* bq = (const float*)d_in[4];
  const float* Wk = (const float*)d_in[5];  const float* bk = (const float*)d_in[6];
  const float* Wv = (const float*)d_in[7];  const float* bv = (const float*)d_in[8];
  const float* Wo = (const float*)d_in[9];  const float* bo = (const float*)d_in[10];
  const float* rec_s  = (const float*)d_in[11];
  const float* bowl_s = (const float*)d_in[12];
  const float* bat_s  = (const float*)d_in[13];

  const size_t xElems = (size_t)Bn * Sn * Dn;  // 4,194,304
  const size_t wElems = (size_t)Dn * Dn;       // 65,536

  char* w = (char*)d_ws;
  u16* xb   = (u16*)w; w += xElems * 2;
  u16* Wqb  = (u16*)w; w += wElems * 2;
  u16* Wkb  = (u16*)w; w += wElems * 2;
  u16* Wvb  = (u16*)w; w += wElems * 2;
  u16* Wob  = (u16*)w; w += wElems * 2;
  u16* Qb   = (u16*)w; w += xElems * 2;
  u16* Kb   = (u16*)w; w += xElems * 2;
  u16* Vt   = (u16*)w; w += xElems * 2;   // [B,H,HD,S]
  u16* aOut = (u16*)w; w += xElems * 2;   // attn@V result, bf16 [B,S,D]

  float* out  = (float*)d_out;
  float* attn = out + xElems;             // [B,H,S,S] fp32 (second output)

  // 1) fp32 -> bf16 conversions
  f32_to_bf16_k<<<2048, 256, 0, stream>>>(x,  xb,  (int)xElems);
  f32_to_bf16_k<<<64,   256, 0, stream>>>(Wq, Wqb, (int)wElems);
  f32_to_bf16_k<<<64,   256, 0, stream>>>(Wk, Wkb, (int)wElems);
  f32_to_bf16_k<<<64,   256, 0, stream>>>(Wv, Wvb, (int)wElems);
  f32_to_bf16_k<<<64,   256, 0, stream>>>(Wo, Wob, (int)wElems);

  // 2) Q/K/V projections (WMMA bf16)
  dim3 gP(Mn / 16, Dn / 16);  // 1024 x 16 tiles
  gemm_proj<<<gP, 32, 0, stream>>>(xb, Wqb, bq, Qb, nullptr, 0);
  gemm_proj<<<gP, 32, 0, stream>>>(xb, Wkb, bk, Kb, nullptr, 0);
  gemm_proj<<<gP, 32, 0, stream>>>(xb, Wvb, bv, Vt, nullptr, 2);

  // 3) scores + per-head bias (lower-triangular tiles only)
  dim3 gS(Sn / 16, Sn / 16, Bn * Hn);  // 128 x 128 x 32
  attn_scores<<<gS, 32, 0, stream>>>(Qb, Kb, bowler, batsman,
                                     rec_s, bowl_s, bat_s, attn);

  // 4) causal softmax in place (fills zeros above diagonal)
  softmax_causal<<<dim3(Sn, Bn * Hn), 256, 0, stream>>>(attn);

  // 5) attn @ V (WMMA, K-loop to the diagonal)
  dim3 gV(Sn / 16, HDn / 16, Bn * Hn);  // 128 x 4 x 32
  attn_v<<<gV, 32, 0, stream>>>(attn, Vt, aOut);

  // 6) output projection -> fp32 d_out
  gemm_proj<<<gP, 32, 0, stream>>>(aOut, Wob, bo, nullptr, out, 1);
}
